// MixFFN_59416577573478
// MI455X (gfx1250) — compile-verified, matrix operand order checked by reading, not run on hardware
//
#include <hip/hip_runtime.h>

// ---------------------------------------------------------------------------
// MixFFN (shared SwiGLU FFN + per-expert rank-16 LoRA, top-2 of 8 experts)
// Restructured so the heavy W2 GEMM runs once on hbar = sum_e ew_e * h_e.
// Heavy GEMMs: bf16 WMMA (v_wmma_f32_16x16x32_bf16) with double-buffered LDS
// staging via gfx1250 async copies (global_load_async_to_lds_b128 / ASYNCcnt).
// ---------------------------------------------------------------------------

#define N_TOK   4096
#define D_MODEL 2048
#define D_FF    8192
#define N_EXP   8
#define LORA_R  16

// GEMM tiling: block = 128 threads = 4 waves (2M x 2N), wave tile 64x32
#define BM 128
#define BN 64
#define BK 64
#define GEMM_THREADS 128
#define LDS_STRIDE 72   // padded bf16 row stride: 36 dwords -> conflict-free b128 reads

typedef __attribute__((ext_vector_type(16))) __bf16       v16bf;
typedef __attribute__((ext_vector_type(8)))  float        v8f;
typedef __attribute__((ext_vector_type(4)))  unsigned int uv4;

__device__ __forceinline__ unsigned short f2bf_bits(float f) {
  union { float f; unsigned u; } v; v.f = f;
  unsigned r = v.u + 0x7FFFu + ((v.u >> 16) & 1u);   // round-to-nearest-even
  return (unsigned short)(r >> 16);
}

// ---------------------------------------------------------------------------
// f32 -> bf16 bulk conversion
// ---------------------------------------------------------------------------
__global__ void __launch_bounds__(256)
cvt_f32_to_bf16(const float* __restrict__ in, unsigned short* __restrict__ out, int n) {
  int i = blockIdx.x * 256 + threadIdx.x;
  if (i < n) out[i] = f2bf_bits(in[i]);
}

// ---------------------------------------------------------------------------
// Routing: logits = x @ gate_w^T, softmax, top-2, renormalize -> ew [N, E]
// ---------------------------------------------------------------------------
__global__ void __launch_bounds__(256)
routing_kernel(const float* __restrict__ x, const float* __restrict__ gw,
               float* __restrict__ ew) {
  const int lane = threadIdx.x & 31;
  const int wave = threadIdx.x >> 5;
  const int row  = blockIdx.x * 8 + wave;
  if (row >= N_TOK) return;

  float acc[N_EXP];
#pragma unroll
  for (int e = 0; e < N_EXP; ++e) acc[e] = 0.f;

  const float* xr = x + (size_t)row * D_MODEL;
  for (int d = lane; d < D_MODEL; d += 32) {
    float xv = xr[d];
#pragma unroll
    for (int e = 0; e < N_EXP; ++e) acc[e] += xv * gw[e * D_MODEL + d];
  }
#pragma unroll
  for (int e = 0; e < N_EXP; ++e) {
#pragma unroll
    for (int off = 16; off > 0; off >>= 1) acc[e] += __shfl_xor(acc[e], off, 32);
  }
  if (lane == 0) {
    float m = acc[0];
#pragma unroll
    for (int e = 1; e < N_EXP; ++e) m = fmaxf(m, acc[e]);
    float p[N_EXP];
#pragma unroll
    for (int e = 0; e < N_EXP; ++e) p[e] = __expf(acc[e] - m);
    int i1 = 0;
#pragma unroll
    for (int e = 1; e < N_EXP; ++e) if (p[e] > p[i1]) i1 = e;
    int i2 = (i1 == 0) ? 1 : 0;
#pragma unroll
    for (int e = 0; e < N_EXP; ++e) if (e != i1 && p[e] > p[i2]) i2 = e;
    float s = p[i1] + p[i2];
#pragma unroll
    for (int e = 0; e < N_EXP; ++e)
      ew[row * N_EXP + e] = (e == i1) ? p[i1] / s : ((e == i2) ? p[i2] / s : 0.f);
  }
}

// ---------------------------------------------------------------------------
// Async global->LDS tile copy (per-lane global_load_async_to_lds_b128).
// Tile is ROWS x 64 bf16, 16B chunks; compile-time trip count -> no exec masks.
// ---------------------------------------------------------------------------
template <int ROWS, int NT>
__device__ __forceinline__ void
async_copy_tile(unsigned short* sdst, const unsigned short* __restrict__ gsrc,
                int ldg, int tid) {
  constexpr int CHUNKS = ROWS * 8;                 // 16B chunks per tile
#pragma unroll
  for (int j = 0; j < CHUNKS / NT; ++j) {
    const int c   = tid + j * NT;
    const int row = c >> 3;
    const int col = (c & 7) << 3;                  // bf16 element offset
    unsigned lds_addr = (unsigned)(size_t)(sdst + row * LDS_STRIDE + col);
    const unsigned short* g = gsrc + (size_t)row * ldg + col;
    asm volatile("global_load_async_to_lds_b128 %0, %1, off"
                 :: "v"(lds_addr), "v"(g) : "memory");
  }
}

// ---------------------------------------------------------------------------
// bf16 WMMA fragment load from LDS, 16x32 tile (ISA 16-bit A/B layout):
// lanes 0-15: row = lane, K chunks {0..7}->VGPR0..3, {16..23}->VGPR4..7
// lanes 16-31: row = lane-16, K chunks {8..15} and {24..31}
// ---------------------------------------------------------------------------
__device__ __forceinline__ v16bf
lds_frag16x32(const unsigned short* p0, int lane) {
  const int r  = lane & 15;
  const int ko = (lane >> 4) << 3;                 // 0 or 8
  const unsigned short* p = p0 + r * LDS_STRIDE + ko;
  union { v16bf v; uv4 q[2]; } u;
  u.q[0] = *(const uv4*)(p);                       // K = ko .. ko+7
  u.q[1] = *(const uv4*)(p + 16);                  // K = ko+16 .. ko+23
  return u.v;
}

// ---------------------------------------------------------------------------
// C[M,N] = A[M,K] @ B[N,K]^T  (A,B bf16 row-major, C f32 row-major)
// Block 128 thr = 4 waves (2Mx2N), wave tile 64x32 (4 A-frags x 2 B-frags).
// Double-buffered LDS stages of BK=64 filled by async copies; one
// s_wait_asynccnt + barrier per stage orders producer/consumer.
// Requires M%128==0, N%64==0, K%64==0 (true for all three GEMMs here).
// ---------------------------------------------------------------------------
__global__ void __launch_bounds__(GEMM_THREADS)
gemm_bf16_nt(const unsigned short* __restrict__ A, const unsigned short* __restrict__ B,
             float* __restrict__ C, int M, int N, int K) {
  __shared__ unsigned short sA[2][BM * LDS_STRIDE];
  __shared__ unsigned short sB[2][BN * LDS_STRIDE];

  const int tid  = threadIdx.x;
  const int lane = tid & 31;
  const int wave = tid >> 5;                       // 0..3
  const int wm   = wave & 1;                       // M direction (64 rows each)
  const int wn   = wave >> 1;                      // N direction (32 cols each)
  const int m0   = blockIdx.y * BM;
  const int n0   = blockIdx.x * BN;

  v8f c0[2], c1[2], c2[2], c3[2];
#pragma unroll
  for (int j = 0; j < 2; ++j) { c0[j] = (v8f){}; c1[j] = (v8f){}; c2[j] = (v8f){}; c3[j] = (v8f){}; }

  // prologue: stage 0
  async_copy_tile<BM, GEMM_THREADS>(sA[0], A + (size_t)m0 * K, K, tid);
  async_copy_tile<BN, GEMM_THREADS>(sB[0], B + (size_t)n0 * K, K, tid);

  const int nst = K / BK;
  for (int s = 0; s < nst; ++s) {
    // own copies for current buffer complete; barrier makes it block-wide and
    // guarantees everyone is done reading the other buffer (prev stage)
    asm volatile("s_wait_asynccnt 0x0" ::: "memory");
    __syncthreads();

    if (s + 1 < nst) {                             // prefetch next stage
      const int kb = (s + 1) * BK;
      async_copy_tile<BM, GEMM_THREADS>(sA[(s + 1) & 1], A + (size_t)m0 * K + kb, K, tid);
      async_copy_tile<BN, GEMM_THREADS>(sB[(s + 1) & 1], B + (size_t)n0 * K + kb, K, tid);
    }

    const unsigned short* at = sA[s & 1] + (wm * 64) * LDS_STRIDE;
    const unsigned short* bt = sB[s & 1] + (wn * 32) * LDS_STRIDE;
#pragma unroll
    for (int ks = 0; ks < 2; ++ks) {
      const int kc = ks * 32;
      v16bf a0 = lds_frag16x32(at + 0  * LDS_STRIDE + kc, lane);
      v16bf a1 = lds_frag16x32(at + 16 * LDS_STRIDE + kc, lane);
      v16bf a2 = lds_frag16x32(at + 32 * LDS_STRIDE + kc, lane);
      v16bf a3 = lds_frag16x32(at + 48 * LDS_STRIDE + kc, lane);
#pragma unroll
      for (int j = 0; j < 2; ++j) {
        v16bf b = lds_frag16x32(bt + (j * 16) * LDS_STRIDE + kc, lane);
        c0[j] = __builtin_amdgcn_wmma_f32_16x16x32_bf16(false, a0, false, b, (short)0, c0[j], false, false);
        c1[j] = __builtin_amdgcn_wmma_f32_16x16x32_bf16(false, a1, false, b, (short)0, c1[j], false, false);
        c2[j] = __builtin_amdgcn_wmma_f32_16x16x32_bf16(false, a2, false, b, (short)0, c2[j], false, false);
        c3[j] = __builtin_amdgcn_wmma_f32_16x16x32_bf16(false, a3, false, b, (short)0, c3[j], false, false);
      }
    }
  }

  // C layout: VGPR v -> row v + 8*(lane>=16); col = lane&15
  const int cn = lane & 15;
  const int mh = (lane >> 4) * 8;
  const int mw = m0 + wm * 64;
  const int nw = n0 + wn * 32;
#pragma unroll
  for (int j = 0; j < 2; ++j) {
    const int nc = nw + j * 16 + cn;
#pragma unroll
    for (int v = 0; v < 8; ++v) {
      C[(size_t)(mw      + mh + v) * N + nc] = c0[j][v];
      C[(size_t)(mw + 16 + mh + v) * N + nc] = c1[j][v];
      C[(size_t)(mw + 32 + mh + v) * N + nc] = c2[j][v];
      C[(size_t)(mw + 48 + mh + v) * N + nc] = c3[j][v];
    }
  }
}

// ---------------------------------------------------------------------------
// LoRA down-projections: U1[e,n,:] = x[n,:] @ A1[e]^T, same for U3.
// ---------------------------------------------------------------------------
__global__ void __launch_bounds__(256)
lora_down_kernel(const float* __restrict__ x, const float* __restrict__ A1,
                 const float* __restrict__ A3, float* __restrict__ U1,
                 float* __restrict__ U3) {
  const int lane = threadIdx.x & 31;
  const int wave = threadIdx.x >> 5;
  const int pair = blockIdx.x * 8 + wave;        // [0, N_TOK * N_EXP)
  const int n = pair >> 3;
  const int e = pair & 7;

  float u1[LORA_R], u3[LORA_R];
#pragma unroll
  for (int r = 0; r < LORA_R; ++r) { u1[r] = 0.f; u3[r] = 0.f; }

  const float* a1 = A1 + (size_t)e * LORA_R * D_MODEL;
  const float* a3 = A3 + (size_t)e * LORA_R * D_MODEL;
  const float* xr = x + (size_t)n * D_MODEL;
  for (int d = lane; d < D_MODEL; d += 32) {
    float xv = xr[d];
#pragma unroll
    for (int r = 0; r < LORA_R; ++r) {
      u1[r] += xv * a1[r * D_MODEL + d];
      u3[r] += xv * a3[r * D_MODEL + d];
    }
  }
#pragma unroll
  for (int r = 0; r < LORA_R; ++r) {
#pragma unroll
    for (int off = 16; off > 0; off >>= 1) {
      u1[r] += __shfl_xor(u1[r], off, 32);
      u3[r] += __shfl_xor(u3[r], off, 32);
    }
  }
  if (lane == 0) {
    float* o1 = U1 + ((size_t)e * N_TOK + n) * LORA_R;
    float* o3 = U3 + ((size_t)e * N_TOK + n) * LORA_R;
#pragma unroll
    for (int r = 0; r < LORA_R; ++r) { o1[r] = u1[r]; o3[r] = u3[r]; }
  }
}

// ---------------------------------------------------------------------------
// Fused expert kernel. One block (256 threads) per token row n.
// For each expert with ew != 0:
//   w1 = C1 + U1 @ B1^T ; w3 = C3 + U3 @ B3^T ; h = silu(w1)*w3 ; wh = ew*h
//   hbar += wh (bf16 out)        T[n,e,:] = wh @ A2[e]^T (deterministic tree)
// ---------------------------------------------------------------------------
__global__ void __launch_bounds__(256)
expert_fuse_kernel(const float* __restrict__ C1, const float* __restrict__ C3,
                   const float* __restrict__ U1, const float* __restrict__ U3,
                   const float* __restrict__ ew, const float* __restrict__ B1,
                   const float* __restrict__ B3, const float* __restrict__ A2,
                   unsigned short* __restrict__ hbar, float* __restrict__ T) {
  const int n    = blockIdx.x;
  const int tid  = threadIdx.x;
  const int lane = tid & 31;
  const int wave = tid >> 5;

  __shared__ float sW[8][LORA_R];

  float hloc[32];
#pragma unroll
  for (int i = 0; i < 32; ++i) hloc[i] = 0.f;

  for (int e = 0; e < N_EXP; ++e) {
    const float we = ew[n * N_EXP + e];           // block-uniform
    float* Tout = T + ((size_t)n * N_EXP + e) * LORA_R;
    if (we == 0.f) {                              // skipped expert: T must be 0
      if (tid < LORA_R) Tout[tid] = 0.f;
      continue;
    }
    float u1[LORA_R], u3[LORA_R];
    const float* u1p = U1 + ((size_t)e * N_TOK + n) * LORA_R;
    const float* u3p = U3 + ((size_t)e * N_TOK + n) * LORA_R;
#pragma unroll
    for (int r = 0; r < LORA_R; ++r) { u1[r] = u1p[r]; u3[r] = u3p[r]; }

    float t[LORA_R];
#pragma unroll
    for (int r = 0; r < LORA_R; ++r) t[r] = 0.f;

    const float* b1 = B1 + (size_t)e * D_FF * LORA_R;
    const float* b3 = B3 + (size_t)e * D_FF * LORA_R;
    const float* a2 = A2 + (size_t)e * LORA_R * D_FF;

    for (int i = 0; i < 32; ++i) {
      const int dd = tid + (i << 8);              // coalesced stride-256
      float w1 = C1[(size_t)n * D_FF + dd];
      float w3 = C3[(size_t)n * D_FF + dd];
      const float* b1r = b1 + (size_t)dd * LORA_R;
      const float* b3r = b3 + (size_t)dd * LORA_R;
#pragma unroll
      for (int r = 0; r < LORA_R; ++r) { w1 += u1[r] * b1r[r]; w3 += u3[r] * b3r[r]; }
      const float h  = (w1 / (1.f + __expf(-w1))) * w3;   // silu(w1) * w3
      const float wh = we * h;
      hloc[i] += wh;
#pragma unroll
      for (int r = 0; r < LORA_R; ++r) t[r] += wh * a2[(size_t)r * D_FF + dd];
    }

    // deterministic block reduction: wave butterfly -> per-wave LDS -> fixed sum
#pragma unroll
    for (int r = 0; r < LORA_R; ++r) {
      float v = t[r];
#pragma unroll
      for (int off = 16; off > 0; off >>= 1) v += __shfl_xor(v, off, 32);
      if (lane == 0) sW[wave][r] = v;
    }
    __syncthreads();
    if (tid < LORA_R) {
      float s = 0.f;
#pragma unroll
      for (int w = 0; w < 8; ++w) s += sW[w][tid];
      Tout[tid] = s;
    }
    __syncthreads();
  }

#pragma unroll
  for (int i = 0; i < 32; ++i)
    hbar[(size_t)n * D_FF + tid + (i << 8)] = f2bf_bits(hloc[i]);
}

// ---------------------------------------------------------------------------
// Rank-16 epilogue: out[n,d] += sum_e sum_r T[n,e,r] * B2[e,d,r]
// ---------------------------------------------------------------------------
__global__ void __launch_bounds__(256)
lora_up_kernel(const float* __restrict__ T, const float* __restrict__ B2,
               float* __restrict__ out) {
  const size_t idx = (size_t)blockIdx.x * 256 + threadIdx.x;
  const int n = (int)(idx / D_MODEL);
  const int d = (int)(idx % D_MODEL);
  float acc = out[idx];
  const float* tn = T + (size_t)n * N_EXP * LORA_R;
#pragma unroll
  for (int e = 0; e < N_EXP; ++e) {
    const float* b2 = B2 + ((size_t)e * D_MODEL + d) * LORA_R;
#pragma unroll
    for (int r = 0; r < LORA_R; ++r) acc += tn[e * LORA_R + r] * b2[r];
  }
  out[idx] = acc;
}

// ---------------------------------------------------------------------------
// Launcher
// ---------------------------------------------------------------------------
extern "C" void kernel_launch(void* const* d_in, const int* in_sizes, int n_in,
                              void* d_out, int out_size, void* d_ws, size_t ws_size,
                              hipStream_t stream) {
  (void)in_sizes; (void)n_in; (void)out_size; (void)ws_size;

  const float* x   = (const float*)d_in[0];
  const float* W1  = (const float*)d_in[1];
  const float* W3  = (const float*)d_in[2];
  const float* W2  = (const float*)d_in[3];
  const float* gw  = (const float*)d_in[4];
  const float* A1  = (const float*)d_in[5];
  const float* B1  = (const float*)d_in[6];
  const float* A3  = (const float*)d_in[7];
  const float* B3  = (const float*)d_in[8];
  const float* A2  = (const float*)d_in[9];
  const float* B2  = (const float*)d_in[10];
  float* out = (float*)d_out;

  // workspace carve-up (~437 MB total)
  char* ws = (char*)d_ws;
  size_t off = 0;
  auto alloc = [&](size_t bytes) -> void* {
    void* p = ws + off;
    off += (bytes + 255) & ~(size_t)255;
    return p;
  };
  unsigned short* xb    = (unsigned short*)alloc((size_t)N_TOK * D_MODEL * 2);
  unsigned short* W1b   = (unsigned short*)alloc((size_t)D_FF * D_MODEL * 2);
  unsigned short* W3b   = (unsigned short*)alloc((size_t)D_FF * D_MODEL * 2);
  unsigned short* W2b   = (unsigned short*)alloc((size_t)D_MODEL * D_FF * 2);
  unsigned short* hbarb = (unsigned short*)alloc((size_t)N_TOK * D_FF * 2);
  float* C1 = (float*)alloc((size_t)N_TOK * D_FF * 4);
  float* C3 = (float*)alloc((size_t)N_TOK * D_FF * 4);
  float* ewp = (float*)alloc((size_t)N_TOK * N_EXP * 4);
  float* U1 = (float*)alloc((size_t)N_EXP * N_TOK * LORA_R * 4);
  float* U3 = (float*)alloc((size_t)N_EXP * N_TOK * LORA_R * 4);
  float* T  = (float*)alloc((size_t)N_TOK * N_EXP * LORA_R * 4);

  // 1) bf16 conversions
  {
    int n = N_TOK * D_MODEL;
    cvt_f32_to_bf16<<<(n + 255) / 256, 256, 0, stream>>>(x, xb, n);
  }
  {
    int n = D_FF * D_MODEL;
    cvt_f32_to_bf16<<<(n + 255) / 256, 256, 0, stream>>>(W1, W1b, n);
    cvt_f32_to_bf16<<<(n + 255) / 256, 256, 0, stream>>>(W3, W3b, n);
    cvt_f32_to_bf16<<<(n + 255) / 256, 256, 0, stream>>>(W2, W2b, n);
  }

  // 2) routing -> ew
  routing_kernel<<<N_TOK / 8, 256, 0, stream>>>(x, gw, ewp);

  // 3) shared projections: C1 = x @ W1^T, C3 = x @ W3^T   [N_TOK, D_FF]
  gemm_bf16_nt<<<dim3(D_FF / BN, N_TOK / BM), GEMM_THREADS, 0, stream>>>(
      xb, W1b, C1, N_TOK, D_FF, D_MODEL);
  gemm_bf16_nt<<<dim3(D_FF / BN, N_TOK / BM), GEMM_THREADS, 0, stream>>>(
      xb, W3b, C3, N_TOK, D_FF, D_MODEL);

  // 4) LoRA down: U1/U3 [E, N_TOK, R]
  lora_down_kernel<<<(N_TOK * N_EXP) / 8, 256, 0, stream>>>(x, A1, A3, U1, U3);

  // 5) fused experts -> hbar (bf16), T [N_TOK, E, R]
  expert_fuse_kernel<<<N_TOK, 256, 0, stream>>>(C1, C3, U1, U3, ewp, B1, B3, A2,
                                                hbarb, T);

  // 6) out = hbar @ W2^T   [N_TOK, D_MODEL]
  gemm_bf16_nt<<<dim3(D_MODEL / BN, N_TOK / BM), GEMM_THREADS, 0, stream>>>(
      hbarb, W2b, out, N_TOK, D_MODEL, D_FF);

  // 7) out += sum_e T_e @ B2[e]^T
  lora_up_kernel<<<(N_TOK * D_MODEL) / 256, 256, 0, stream>>>(T, B2, out);
}
